// std_GCN_31619549233339
// MI455X (gfx1250) — matrix operand order repaired and checked by reading, not compiled, not run on hardware
//
#include <hip/hip_runtime.h>

// CDNA5 / gfx1250 GCN forward. GEMMs use v_wmma_f32_16x16x32_f16 (wave32),
// edge aggregation uses vectorized global float atomics (L2-resident).

typedef __attribute__((ext_vector_type(16))) _Float16 v16h;
typedef __attribute__((ext_vector_type(8)))  float    v8f;

union FragH { v16h v; uint4 q[2]; };

#define D 64

// ---------------- utility ----------------
__global__ void fill_f32_k(float* __restrict__ p, float v, int n) {
  int i = blockIdx.x * blockDim.x + threadIdx.x;
  if (i < n) p[i] = v;
}

// x0 = feat / rowsum(feat), stored as f16 (GEMM A operand precision)
__global__ void normalize_feat_k(const float* __restrict__ feat,
                                 _Float16* __restrict__ xh, int N) {
  int gt = blockIdx.x * blockDim.x + threadIdx.x;
  int node = gt >> 5;
  int lane = gt & 31;
  if (node >= N) return;
  const float* row = feat + (size_t)node * D;
  float2 v = *(const float2*)(row + lane * 2);
  float s = v.x + v.y;
  #pragma unroll
  for (int off = 16; off > 0; off >>= 1) s += __shfl_xor(s, off, 32);
  float inv = 1.0f / s;
  _Float16* o = xh + (size_t)node * D + lane * 2;
  o[0] = (_Float16)(v.x * inv);
  o[1] = (_Float16)(v.y * inv);
}

// deg_row = segment_sum(edge_vals by src)
__global__ void deg_row_k(const int* __restrict__ src, const float* __restrict__ ev,
                          float* __restrict__ degrow, int E) {
  int e = blockIdx.x * blockDim.x + threadIdx.x;
  if (e < E) atomicAdd(&degrow[src[e]], ev[e]);
}

// w_hat = ev / (deg_row[src]*deg_row[dst])  (guarded); deg[dst] += w_hat
// (deg buffer pre-initialized to 1.0 == self-loop weight)
__global__ void what_deg_k(const int* __restrict__ src, const int* __restrict__ dst,
                           const float* __restrict__ ev, const float* __restrict__ degrow,
                           float* __restrict__ wbuf, float* __restrict__ deg, int E) {
  int e = blockIdx.x * blockDim.x + threadIdx.x;
  if (e >= E) return;
  float ds = degrow[src[e]]; ds = ds > 0.0f ? ds : 1.0f;
  float dd = degrow[dst[e]]; dd = dd > 0.0f ? dd : 1.0f;
  float wh = ev[e] / (ds * dd);
  wbuf[e] = wh;
  atomicAdd(&deg[dst[e]], wh);
}

__global__ void rsqrt_k(float* __restrict__ p, int n) {
  int i = blockIdx.x * blockDim.x + threadIdx.x;
  if (i < n) { float d = p[i]; p[i] = d > 0.0f ? rsqrtf(d) : 0.0f; }
}

__global__ void norm_fin_k(const int* __restrict__ src, const int* __restrict__ dst,
                           const float* __restrict__ dinv, float* __restrict__ wbuf, int E) {
  int e = blockIdx.x * blockDim.x + threadIdx.x;
  if (e < E) wbuf[e] = dinv[src[e]] * wbuf[e] * dinv[dst[e]];
}

// pack W (f32 64x64) into per-lane B fragments (f16), layout:
// Wp[((layer*8 + kt*4 + nt)*32 + lane)*16 + j] = W[k][n]
//   lanes 0-15: n = nt*16+lane,   k = kt*32 + (j&7) + (j>>3)*16
//   lanes16-31: n = nt*16+lane-16, k = kt*32 + 8 + (j&7) + (j>>3)*16
__global__ void pack_w_k(const float* __restrict__ W1, const float* __restrict__ W2,
                         _Float16* __restrict__ Wp) {
  int t = blockIdx.x * blockDim.x + threadIdx.x;
  if (t >= 2 * 8 * 32 * 16) return;
  int j     = t & 15;
  int lane  = (t >> 4) & 31;
  int frag  = (t >> 9) & 7;   // kt*4+nt
  int layer = t >> 12;
  int nt = frag & 3;
  int kt = frag >> 2;
  int col = nt * 16 + (lane & 15);
  int k   = kt * 32 + ((lane >> 4) << 3) + (j & 7) + ((j >> 3) << 4);
  const float* W = layer ? W2 : W1;
  Wp[t] = (_Float16)W[k * D + col];
}

// H[N,64] = X[N,64](f16) @ Wp(layer), one wave per 16-row tile, 8 WMMAs/wave.
__global__ __launch_bounds__(256) void gemm64_wmma_k(const _Float16* __restrict__ X,
                                                     const _Float16* __restrict__ Wp,
                                                     float* __restrict__ H, int mTiles) {
  int gt = blockIdx.x * blockDim.x + threadIdx.x;
  int wave = gt >> 5;
  int lane = gt & 31;
  if (wave >= mTiles) return;          // wave-uniform: EXEC all-ones inside
  int hi = lane >> 4;                  // 0: K {0..7,16..23}; 1: K {8..15,24..31}
  int lm = lane & 15;
  int m  = wave * 16 + lm;
  int klo = hi * 8;
  const _Float16* xr = X + (size_t)m * D;
  FragH a0, a1;
  a0.q[0] = *(const uint4*)(xr + klo);
  a0.q[1] = *(const uint4*)(xr + klo + 16);
  a1.q[0] = *(const uint4*)(xr + 32 + klo);
  a1.q[1] = *(const uint4*)(xr + 32 + klo + 16);
  v8f z = {};
  v8f acc[4] = {z, z, z, z};
  const v16h* B = (const v16h*)Wp;
  #pragma unroll
  for (int kt = 0; kt < 2; ++kt) {
    v16h a = kt ? a1.v : a0.v;
    #pragma unroll
    for (int nt = 0; nt < 4; ++nt) {
      v16h b = B[(kt * 4 + nt) * 32 + lane];
      acc[nt] = __builtin_amdgcn_wmma_f32_16x16x32_f16(
          false, a, false, b, (short)0, acc[nt], false, false);
    }
  }
  int rbase = wave * 16 + hi * 8;      // D layout: VGPR r -> rows r / r+8
  #pragma unroll
  for (int nt = 0; nt < 4; ++nt) {
    int col = nt * 16 + lm;
    #pragma unroll
    for (int r = 0; r < 8; ++r)
      H[(size_t)(rbase + r) * D + col] = acc[nt][r];
  }
}

// agg[i,c] = b[c] + dinv[i]^2 * H[i,c]   (self-loop term; initializes buffer)
__global__ void agg_init_k(const float* __restrict__ H, const float* __restrict__ dinv,
                           const float* __restrict__ bias, float* __restrict__ AGG, int N) {
  int idx = blockIdx.x * blockDim.x + threadIdx.x;
  if (idx >= N * D) return;
  int i = idx >> 6, c = idx & 63;
  float di = dinv[i];
  AGG[idx] = bias[c] + di * di * H[idx];
}

// agg[dst,:] += norm_e * H[src,:], 16 threads/edge, float4 gather + 4 atomics
__global__ void edge_scatter_vec_k(const int* __restrict__ src, const int* __restrict__ dst,
                                   const float* __restrict__ norm, const float* __restrict__ H,
                                   float* __restrict__ AGG, int E) {
  long long t = (long long)blockIdx.x * blockDim.x + threadIdx.x;
  if (t >= (long long)E * 16) return;
  int e  = (int)(t >> 4);
  int c0 = ((int)t & 15) << 2;
  float nv = norm[e];
  int s = src[e], d = dst[e];
  float4 h4 = *(const float4*)(H + (size_t)s * D + c0);
  float* o = AGG + (size_t)d * D + c0;
  atomicAdd(o + 0, nv * h4.x);
  atomicAdd(o + 1, nv * h4.y);
  atomicAdd(o + 2, nv * h4.z);
  atomicAdd(o + 3, nv * h4.w);
}

// x_{l+1} = relu(agg) stored as f16 for next GEMM
__global__ void relu_cvt_k(const float* __restrict__ AGG, _Float16* __restrict__ xh, int N) {
  int idx = blockIdx.x * blockDim.x + threadIdx.x;
  if (idx < N * D) xh[idx] = (_Float16)fmaxf(AGG[idx], 0.0f);
}

// layer 3: h3[i] = relu(agg2[i,:]) . W3 ; out[i] = b3 + dinv^2 * h3[i]
__global__ void final_layer_k(const float* __restrict__ AGG, const float* __restrict__ dinv,
                              const float* __restrict__ W3, const float* __restrict__ b3,
                              float* __restrict__ h3, float* __restrict__ out, int N) {
  int gt = blockIdx.x * blockDim.x + threadIdx.x;
  int node = gt >> 5;
  int lane = gt & 31;
  if (node >= N) return;
  float2 a = *(const float2*)(AGG + (size_t)node * D + lane * 2);
  float2 w = *(const float2*)(W3 + lane * 2);
  float s = fmaxf(a.x, 0.0f) * w.x + fmaxf(a.y, 0.0f) * w.y;
  #pragma unroll
  for (int off = 16; off > 0; off >>= 1) s += __shfl_xor(s, off, 32);
  if (lane == 0) {
    float di = dinv[node];
    h3[node] = s;
    out[node] = b3[0] + di * di * s;
  }
}

__global__ void edge_scatter1_k(const int* __restrict__ src, const int* __restrict__ dst,
                                const float* __restrict__ norm, const float* __restrict__ h3,
                                float* __restrict__ out, int E) {
  int e = blockIdx.x * blockDim.x + threadIdx.x;
  if (e < E) atomicAdd(&out[dst[e]], norm[e] * h3[src[e]]);
}

// ---------------- launch ----------------
extern "C" void kernel_launch(void* const* d_in, const int* in_sizes, int n_in,
                              void* d_out, int out_size, void* d_ws, size_t ws_size,
                              hipStream_t stream) {
  const float* feat = (const float*)d_in[0];
  const int*   eidx = (const int*)d_in[1];
  const float* ev   = (const float*)d_in[2];
  const float* W1 = (const float*)d_in[3];
  const float* b1 = (const float*)d_in[4];
  const float* W2 = (const float*)d_in[5];
  const float* b2 = (const float*)d_in[6];
  const float* W3 = (const float*)d_in[7];
  const float* b3 = (const float*)d_in[8];
  float* out = (float*)d_out;

  const int E = in_sizes[2];
  const int N = in_sizes[0] / D;       // 100000 = 6250 * 16
  const int* src = eidx;
  const int* dst = eidx + E;

  // workspace carve-up (256B aligned)
  char* ws = (char*)d_ws;
  size_t off = 0;
  auto take = [&](size_t bytes) -> void* {
    void* p = ws + off;
    off += (bytes + 255) & ~(size_t)255;
    return p;
  };
  _Float16* xh     = (_Float16*)take((size_t)N * D * 2);
  float*    H      = (float*)take((size_t)N * D * 4);
  float*    AGG    = (float*)take((size_t)N * D * 4);
  float*    degrow = (float*)take((size_t)N * 4);
  float*    dinv   = (float*)take((size_t)N * 4);   // deg accumulator -> dinv
  float*    wbuf   = (float*)take((size_t)E * 4);   // w_hat -> norm
  float*    h3     = (float*)take((size_t)N * 4);
  _Float16* Wp     = (_Float16*)take(2 * 8 * 32 * 16 * 2);

  const int TB = 256;
  int bN   = (N + TB - 1) / TB;
  int bE   = (E + TB - 1) / TB;
  int bND  = (N * D + TB - 1) / TB;
  int bNw  = ((N * 32) + TB - 1) / TB;              // wave-per-node kernels
  int mTiles = N / 16;
  int bGemm = (mTiles * 32 + TB - 1) / TB;
  int bScat = (int)(((long long)E * 16 + TB - 1) / TB);

  // normalization setup
  fill_f32_k<<<bN, TB, 0, stream>>>(degrow, 0.0f, N);
  fill_f32_k<<<bN, TB, 0, stream>>>(dinv, 1.0f, N);   // self-loop weight
  normalize_feat_k<<<bNw, TB, 0, stream>>>(feat, xh, N);
  deg_row_k<<<bE, TB, 0, stream>>>(src, ev, degrow, E);
  what_deg_k<<<bE, TB, 0, stream>>>(src, dst, ev, degrow, wbuf, dinv, E);
  rsqrt_k<<<bN, TB, 0, stream>>>(dinv, N);
  norm_fin_k<<<bE, TB, 0, stream>>>(src, dst, dinv, wbuf, E);
  pack_w_k<<<(2 * 8 * 32 * 16 + TB - 1) / TB, TB, 0, stream>>>(W1, W2, Wp);

  // layer 1
  gemm64_wmma_k<<<bGemm, TB, 0, stream>>>(xh, Wp, H, mTiles);
  agg_init_k<<<bND, TB, 0, stream>>>(H, dinv, b1, AGG, N);
  edge_scatter_vec_k<<<bScat, TB, 0, stream>>>(src, dst, wbuf, H, AGG, E);
  relu_cvt_k<<<bND, TB, 0, stream>>>(AGG, xh, N);

  // layer 2
  gemm64_wmma_k<<<bGemm, TB, 0, stream>>>(xh, Wp + 8 * 32 * 16, H, mTiles);
  agg_init_k<<<bND, TB, 0, stream>>>(H, dinv, b2, AGG, N);
  edge_scatter_vec_k<<<bScat, TB, 0, stream>>>(src, dst, wbuf, H, AGG, E);

  // layer 3 (64 -> 1)
  final_layer_k<<<bNw, TB, 0, stream>>>(AGG, dinv, W3, b3, h3, out, N);
  edge_scatter1_k<<<bE, TB, 0, stream>>>(src, dst, wbuf, h3, out, E);
}
